// BitLinear158_88905823027952
// MI455X (gfx1250) — 8-bit matrix operand order repaired and checked by reading, compiled, not run on hardware
//
#include <hip/hip_runtime.h>
#include <stdint.h>

typedef uint32_t u32;
typedef unsigned long long u64;
typedef unsigned int v4u __attribute__((ext_vector_type(4)));
typedef int          v8i __attribute__((ext_vector_type(8)));
typedef int          v4i __attribute__((ext_vector_type(4)));

#define D_IN   4096
#define D_OUT  4096
#define M_TOT  8192
#define NGROUP 64
#define EPSQ   1e-5f

__global__ void __launch_bounds__(256) act_quant_kernel(
    const float* __restrict__ x, int8_t* __restrict__ qx, float* __restrict__ invs)
{
    const int token = blockIdx.x;
    const int wave  = threadIdx.x >> 5;
    const int lane  = threadIdx.x & 31;

    #pragma unroll
    for (int gi = 0; gi < 8; ++gi) {
        const int g = gi * 8 + wave;
        const float2 v = ((const float2*)(x + (size_t)token * D_IN + g * 64))[lane];
        float am = fmaxf(fabsf(v.x), fabsf(v.y));
        #pragma unroll
        for (int off = 16; off > 0; off >>= 1)
            am = fmaxf(am, __shfl_xor(am, off, 32));
        am = fmaxf(am, EPSQ);
        const float scale = 127.0f / am;
        int qa = (int)rintf(v.x * scale); qa = max(-127, min(127, qa));
        int qb = (int)rintf(v.y * scale); qb = max(-127, min(127, qb));
        const uint16_t packed = (uint16_t)((qa & 0xFF) | ((qb & 0xFF) << 8));
        ((uint16_t*)(qx + (size_t)token * D_IN + g * 64))[lane] = packed;
        if (lane == 0) invs[(size_t)token * NGROUP + g] = am / 127.0f;
    }
}

__global__ void __launch_bounds__(256) w_quant_kernel(
    const float* __restrict__ w, int8_t* __restrict__ t, float* __restrict__ ws)
{
    __shared__ float red[256];
    const int row = blockIdx.x;
    const float* wr = w + (size_t)row * D_IN;

    float s = 0.0f;
    for (int i = threadIdx.x; i < D_IN; i += 256) s += fabsf(wr[i]);
    red[threadIdx.x] = s;
    __syncthreads();
    for (int k = 128; k > 0; k >>= 1) {
        if (threadIdx.x < k) red[threadIdx.x] += red[threadIdx.x + k];
        __syncthreads();
    }
    const float sm = fmaxf(red[0] * (1.0f / D_IN), EPSQ);
    if (threadIdx.x == 0) ws[row] = sm;
    const float rinv = 1.0f / sm;
    for (int i = threadIdx.x; i < D_IN; i += 256) {
        int q = (int)rintf(wr[i] * rinv);
        q = max(-1, min(1, q));
        t[(size_t)row * D_IN + i] = (int8_t)q;
    }
}

__device__ __forceinline__ void tdm_load_2d(u32 lds_addr, const void* gptr,
                                            u32 tensor_d0, u32 tensor_d1,
                                            u32 tile_d0, u32 tile_d1,
                                            u64 stride0, u32 ds_code)
{
    const u64 ga = (u64)(uintptr_t)gptr;
    v4u g0;
    g0.x = 1u;
    g0.y = lds_addr;
    g0.z = (u32)ga;
    g0.w = (u32)(ga >> 32) | (2u << 30);
    v8i g1;
    g1[0] = (int)(ds_code << 16);
    g1[1] = (int)((tensor_d0 & 0xFFFFu) << 16);
    g1[2] = (int)((tensor_d0 >> 16) | ((tensor_d1 & 0xFFFFu) << 16));
    g1[3] = (int)((tensor_d1 >> 16) | (tile_d0 << 16));
    g1[4] = (int)(tile_d1 & 0xFFFFu);
    g1[5] = (int)(u32)(stride0 & 0xFFFFFFFFu);
    g1[6] = (int)(u32)((stride0 >> 32) & 0xFFFFu);
    g1[7] = 0;
    const v4i z4 = {};
    const v8i z8 = {};
    __builtin_amdgcn_tensor_load_to_lds(g0, g1, z4, z4, z8, 0);
}

__global__ void __launch_bounds__(256) bitgemm_kernel(
    const int8_t* __restrict__ qx, const float* __restrict__ invs,
    const int8_t* __restrict__ wt, const float* __restrict__ ws,
    float* __restrict__ out)
{
    __shared__ __align__(16) int8_t As[3][64 * 64];
    __shared__ __align__(16) int8_t Bs[3][256 * 64];
    __shared__ __align__(16) float  Ss[64 * NGROUP];

    const int tid    = threadIdx.x;
    const int lane   = tid & 31;
    const int waveId = tid >> 5;
    const int hi     = lane >> 4;
    const int lo     = lane & 15;
    const int wm     = waveId & 1;
    const int wn     = waveId >> 1;

    const int Mblk = blockIdx.y * 64;
    const int Nblk = blockIdx.x * 256;

    float acc[2][4][8];
    #pragma unroll
    for (int a = 0; a < 2; ++a)
        #pragma unroll
        for (int b = 0; b < 4; ++b)
            #pragma unroll
            for (int v = 0; v < 8; ++v) acc[a][b][v] = 0.0f;

    if (tid < 32) {
        tdm_load_2d((u32)(uintptr_t)(void*)&Ss[0], invs + (size_t)Mblk * NGROUP,
                    (u32)M_TOT * NGROUP, 1u, 4096u, 1u, 4096u, 2u);
        #pragma unroll
        for (int p = 0; p < 2; ++p) {
            tdm_load_2d((u32)(uintptr_t)(void*)&As[p][0],
                        qx + (size_t)Mblk * D_IN + (size_t)p * 64,
                        D_IN, M_TOT, 64u, 64u, D_IN, 0u);
            tdm_load_2d((u32)(uintptr_t)(void*)&Bs[p][0],
                        wt + (size_t)Nblk * D_IN + (size_t)p * 64,
                        D_IN, D_OUT, 64u, 256u, D_IN, 0u);
        }
    }

    const v8i cz = {};
    for (int g = 0; g < NGROUP; ++g) {
        if (tid < 32) {
            if (g + 1 < NGROUP) __builtin_amdgcn_s_wait_tensorcnt(2);
            else                __builtin_amdgcn_s_wait_tensorcnt(0);
        }
        __syncthreads();
        if ((g + 2 < NGROUP) && tid < 32) {
            const int nb = (g + 2) % 3;
            tdm_load_2d((u32)(uintptr_t)(void*)&As[nb][0],
                        qx + (size_t)Mblk * D_IN + (size_t)(g + 2) * 64,
                        D_IN, M_TOT, 64u, 64u, D_IN, 0u);
            tdm_load_2d((u32)(uintptr_t)(void*)&Bs[nb][0],
                        wt + (size_t)Nblk * D_IN + (size_t)(g + 2) * 64,
                        D_IN, D_OUT, 64u, 256u, D_IN, 0u);
        }
        const int cb = g % 3;

        v8i bfrag[4];
        #pragma unroll
        for (int nt = 0; nt < 4; ++nt) {
            const int n = wn * 64 + nt * 16 + lo;
            const int8_t* bp = &Bs[cb][n * 64];
            union { u64 q[4]; v8i v; } u;
            u.q[0] = *(const u64*)(bp + hi * 16 + 0);
            u.q[1] = *(const u64*)(bp + hi * 16 + 8);
            u.q[2] = *(const u64*)(bp + 32 + hi * 16 + 0);
            u.q[3] = *(const u64*)(bp + 32 + hi * 16 + 8);
            bfrag[nt] = u.v;
        }
        v8i afrag[2];
        float sv[2][8];
        #pragma unroll
        for (int mt = 0; mt < 2; ++mt) {
            const int m = wm * 32 + mt * 16 + lo;
            const int8_t* ap = &As[cb][m * 64];
            union { u64 q[4]; v8i v; } u;
            u.q[0] = *(const u64*)(ap + hi * 16 + 0);
            u.q[1] = *(const u64*)(ap + hi * 16 + 8);
            u.q[2] = *(const u64*)(ap + 32 + hi * 16 + 0);
            u.q[3] = *(const u64*)(ap + 32 + hi * 16 + 8);
            afrag[mt] = u.v;
            const int rbase = wm * 32 + mt * 16 + 8 * hi;
            #pragma unroll
            for (int v = 0; v < 8; ++v)
                sv[mt][v] = Ss[(rbase + v) * NGROUP + g];
        }

        #pragma unroll
        for (int mt = 0; mt < 2; ++mt) {
            #pragma unroll
            for (int nt = 0; nt < 4; ++nt) {
                v8i d = __builtin_amdgcn_wmma_i32_16x16x64_iu8(
                     true, afrag[mt],  true, bfrag[nt],
                    cz,  false,  false);
                #pragma unroll
                for (int v = 0; v < 8; ++v)
                    acc[mt][nt][v] += (float)d[v] * sv[mt][v];
            }
        }
    }

    #pragma unroll
    for (int nt = 0; nt < 4; ++nt) {
        const int col = Nblk + wn * 64 + nt * 16 + lo;
        const float wsc = ws[col];
        #pragma unroll
        for (int mt = 0; mt < 2; ++mt) {
            const int rbase = Mblk + wm * 32 + mt * 16 + 8 * hi;
            #pragma unroll
            for (int v = 0; v < 8; ++v)
                out[(size_t)(rbase + v) * D_OUT + col] = acc[mt][nt][v] * wsc;
        }
    }
}

extern "C" void kernel_launch(void* const* d_in, const int* in_sizes, int n_in,
                              void* d_out, int out_size, void* d_ws, size_t ws_size,
                              hipStream_t stream)
{
    const float* x = (const float*)d_in[0];
    const float* w = (const float*)d_in[1];
    float* out = (float*)d_out;

    const size_t QX_BYTES = (size_t)M_TOT * D_IN;
    const size_t T_BYTES  = (size_t)D_OUT * D_IN;
    int8_t* qx   = (int8_t*)d_ws;
    int8_t* twt  = qx + QX_BYTES;
    float*  invs = (float*)(twt + T_BYTES);
    float*  wsc  = invs + (size_t)M_TOT * NGROUP;

    act_quant_kernel<<<M_TOT, 256, 0, stream>>>(x, qx, invs);
    w_quant_kernel<<<D_OUT, 256, 0, stream>>>(w, twt, wsc);

    dim3 grid(D_OUT / 256, M_TOT / 64);
    bitgemm_kernel<<<grid, 256, 0, stream>>>(qx, invs, twt, wsc, out);
}
